// VectorQuantizer_69896297775564
// MI455X (gfx1250) — matrix operand order repaired and checked by reading, not compiled, not run on hardware
//
#include <hip/hip_runtime.h>

typedef __attribute__((ext_vector_type(16))) _Float16 v16h;
typedef __attribute__((ext_vector_type(8)))  float    v8f;

#define VQ_N  131072
#define VQ_D  64
#define VQ_K  1024
#define VQ_ND (VQ_N * VQ_D)
#define LDS_STRIDE 72   // halfs per codebook row in LDS (64 + 8 pad, spreads banks)

#if __has_builtin(__builtin_amdgcn_sched_barrier)
#define SCHED_FENCE() __builtin_amdgcn_sched_barrier(0)
#else
#define SCHED_FENCE()
#endif

// ---------------------------------------------------------------------------
// Kernel 1: codebook -> f16 copy + per-code squared norms (f32)
// ---------------------------------------------------------------------------
__global__ void vq_prep_kernel(const float* __restrict__ cb,
                               _Float16* __restrict__ cb16,
                               float* __restrict__ cnorm) {
    int k = blockIdx.x * blockDim.x + threadIdx.x;
    if (k < VQ_K) {
        float s = 0.0f;
        #pragma unroll 8
        for (int d = 0; d < VQ_D; ++d) {
            float v = cb[k * VQ_D + d];
            s += v * v;
            cb16[k * VQ_D + d] = (_Float16)v;
        }
        cnorm[k] = s;
    }
}

// ---------------------------------------------------------------------------
// Helpers for the WMMA argmin kernel
// ---------------------------------------------------------------------------
__device__ __forceinline__ void vq_load_b(const _Float16* p, v16h& b0, v16h& b1) {
    __builtin_memcpy(&b0, p, 32);        // K-slice 0..31 for this lane (2 x b128)
    __builtin_memcpy(&b1, p + 32, 32);   // K-slice 32..63 for this lane
}

__device__ __forceinline__ void vq_update(const v8f& acc, float cn, int col,
                                          float* minv, int* mini) {
    #pragma unroll
    for (int r = 0; r < 8; ++r) {
        float score = __builtin_fmaf(-2.0f, acc[r], cn);   // ||e||^2 - 2 x.e
        if (score < minv[r]) { minv[r] = score; mini[r] = col; }
    }
}

// ---------------------------------------------------------------------------
// Kernel 2: WMMA distance GEMM + fused argmin.
// Block = 256 threads = 8 waves; block handles 128 rows, full K=1024 columns.
// Whole f16 codebook staged in LDS (144KB padded) + norms (4KB).
// Software pipeline enforced with sched_barriers:
//   [WMMA tile t] -> [DS issue tile t+2] -> [argmin epilogue tile t]
// so each WMMA's s_wait_dscnt is satisfied by the time it issues.
// ---------------------------------------------------------------------------
__launch_bounds__(256)
__global__ void vq_argmin_kernel(const float* __restrict__ x,
                                 const _Float16* __restrict__ cb16,
                                 const float* __restrict__ cnorm,
                                 int* __restrict__ idx_out,
                                 float* __restrict__ idx_out_f) {
    extern __shared__ char smem[];
    _Float16* scb   = (_Float16*)smem;                               // [K][LDS_STRIDE]
    float*    snorm = (float*)(smem + VQ_K * LDS_STRIDE * 2);        // [K]

    const int tid = threadIdx.x;

    // Cooperative stage of f16 codebook into padded LDS (16B vector copies).
    for (int k = tid; k < VQ_K; k += 256) {
        const uint4* src = (const uint4*)(cb16 + k * VQ_D);   // 64 halfs = 8 x 16B
        uint4*       dst = (uint4*)(scb + k * LDS_STRIDE);    // 144B row stride
        #pragma unroll
        for (int j = 0; j < 8; ++j) dst[j] = src[j];
    }
    for (int k = tid; k < VQ_K; k += 256) snorm[k] = cnorm[k];
    __syncthreads();

    const int wave  = tid >> 5;
    const int lane  = tid & 31;
    const int lhalf = (lane >> 4) & 1;   // which 16-lane half
    const int l15   = lane & 15;
    const int row   = blockIdx.x * 128 + wave * 16 + l15;

    // A tile: 16x64 f16, per k-step (contiguous k, matches B):
    //   lanes 0-15  : halfs 0..15 = K base+0..15
    //   lanes 16-31 : halfs 0..15 = K base+16..31
    v16h a[2];
    {
        const float* xp = x + (size_t)row * VQ_D;
        #pragma unroll
        for (int s = 0; s < 2; ++s) {
            const int lo = s * 32 + lhalf * 16;
            _Float16 ta[16];
            #pragma unroll
            for (int j = 0; j < 8; ++j) ta[j]     = (_Float16)xp[lo + j];
            #pragma unroll
            for (int j = 0; j < 8; ++j) ta[8 + j] = (_Float16)xp[lo + 8 + j];
            __builtin_memcpy(&a[s], ta, 32);
        }
    }

    float minv[8];
    int   mini[8];
    #pragma unroll
    for (int r = 0; r < 8; ++r) { minv[r] = 3.402823e38f; mini[r] = 0; }

    // B tile: 32x16 f16 per k-step, contiguous k (matches A):
    //   lanes 0-15  : halfs 0..15 = K base+0..15   (col = lane)
    //   lanes 16-31 : halfs 0..15 = K base+16..31  (col = lane-16)
    const _Float16* bbase = scb + l15 * LDS_STRIDE + lhalf * 16;
    const int cstep = 16 * LDS_STRIDE;   // halfs between column tiles

    // Prologue: fill both pipeline buffers (tiles 0 and 1).
    v16h b0a, b1a, b0b, b1b;
    vq_load_b(bbase,         b0a, b1a);
    vq_load_b(bbase + cstep, b0b, b1b);
    float cna = snorm[l15];
    float cnb = snorm[16 + l15];

    for (int ct = 0; ct < 64; ct += 2) {
        // ---- stage 1: WMMA on tile ct (buffer A) ----
        v8f acc = {};
        acc = __builtin_amdgcn_wmma_f32_16x16x32_f16(false, a[0], false, b0a,
                                                     (short)0, acc, false, false);
        acc = __builtin_amdgcn_wmma_f32_16x16x32_f16(false, a[1], false, b1a,
                                                     (short)0, acc, false, false);
        SCHED_FENCE();
        // ---- stage 2: issue LDS loads for tile ct+2 into buffer A ----
        float ncn_a;
        {
            int nt = ct + 2; if (nt > 63) nt = 63;        // clamped prefetch
            vq_load_b(bbase + nt * cstep, b0a, b1a);
            ncn_a = snorm[nt * 16 + l15];
        }
        SCHED_FENCE();
        // ---- stage 3: argmin epilogue for tile ct (hides the loads) ----
        vq_update(acc, cna, ct * 16 + l15, minv, mini);
        cna = ncn_a;
        SCHED_FENCE();

        // ---- stage 1': WMMA on tile ct+1 (buffer B) ----
        v8f acc2 = {};
        acc2 = __builtin_amdgcn_wmma_f32_16x16x32_f16(false, a[0], false, b0b,
                                                      (short)0, acc2, false, false);
        acc2 = __builtin_amdgcn_wmma_f32_16x16x32_f16(false, a[1], false, b1b,
                                                      (short)0, acc2, false, false);
        SCHED_FENCE();
        // ---- stage 2': issue LDS loads for tile ct+3 into buffer B ----
        float ncn_b;
        {
            int nt = ct + 3; if (nt > 63) nt = 63;        // clamped prefetch
            vq_load_b(bbase + nt * cstep, b0b, b1b);
            ncn_b = snorm[nt * 16 + l15];
        }
        SCHED_FENCE();
        // ---- stage 3': argmin epilogue for tile ct+1 ----
        vq_update(acc2, cnb, (ct + 1) * 16 + l15, minv, mini);
        cnb = ncn_b;
        SCHED_FENCE();
    }

    // Reduce argmin across the 16 lanes that share each output row.
    #pragma unroll
    for (int m = 1; m < 16; m <<= 1) {
        #pragma unroll
        for (int r = 0; r < 8; ++r) {
            float ov = __shfl_xor(minv[r], m, 32);
            int   oi = __shfl_xor(mini[r], m, 32);
            if (ov < minv[r] || (ov == minv[r] && oi < mini[r])) {
                minv[r] = ov; mini[r] = oi;
            }
        }
    }

    if (l15 == 0) {
        #pragma unroll
        for (int r = 0; r < 8; ++r) {
            int n = blockIdx.x * 128 + wave * 16 + r + lhalf * 8;
            idx_out[n]   = mini[r];
            idx_out_f[n] = (float)mini[r];
        }
    }
}

// ---------------------------------------------------------------------------
// Kernel 3: gather quantized vectors, write straight-through output,
// deterministic per-block partial sums of (q - x)^2.
// ---------------------------------------------------------------------------
__launch_bounds__(256)
__global__ void vq_gather_kernel(const float* __restrict__ x,
                                 const float* __restrict__ cb,
                                 const int* __restrict__ idx,
                                 float* __restrict__ out_q,
                                 float* __restrict__ partials) {
    __shared__ float red[256];
    const int e = blockIdx.x * 256 + threadIdx.x;   // grid covers ND exactly
    const int n = e >> 6;
    const int d = e & 63;
    const float q  = cb[idx[n] * VQ_D + d];
    const float xv = x[e];
    out_q[e] = xv + (q - xv);                       // straight-through value
    const float diff = q - xv;
    red[threadIdx.x] = diff * diff;
    __syncthreads();
    #pragma unroll
    for (int s = 128; s > 0; s >>= 1) {
        if (threadIdx.x < s) red[threadIdx.x] += red[threadIdx.x + s];
        __syncthreads();
    }
    if (threadIdx.x == 0) partials[blockIdx.x] = red[0];
}

// ---------------------------------------------------------------------------
// Kernel 4: deterministic final reduction -> loss scalar.
// ---------------------------------------------------------------------------
__launch_bounds__(256)
__global__ void vq_loss_kernel(const float* __restrict__ partials,
                               float* __restrict__ out_loss) {
    __shared__ float red[256];
    float s = 0.0f;
    for (int j = threadIdx.x; j < (VQ_ND / 256); j += 256) s += partials[j];
    red[threadIdx.x] = s;
    __syncthreads();
    #pragma unroll
    for (int k = 128; k > 0; k >>= 1) {
        if (threadIdx.x < k) red[threadIdx.x] += red[threadIdx.x + k];
        __syncthreads();
    }
    if (threadIdx.x == 0) {
        float mean = red[0] / (float)VQ_ND;
        out_loss[0] = mean + 0.25f * mean;   // q_latent + commitment * e_latent
    }
}

// ---------------------------------------------------------------------------
extern "C" void kernel_launch(void* const* d_in, const int* in_sizes, int n_in,
                              void* d_out, int out_size, void* d_ws, size_t ws_size,
                              hipStream_t stream) {
    (void)in_sizes; (void)n_in; (void)out_size; (void)ws_size;
    const float* x  = (const float*)d_in[0];   // [N, D] f32
    const float* cb = (const float*)d_in[1];   // [K, D] f32

    char* ws = (char*)d_ws;
    float*    cnorm    = (float*)ws;                       // K floats      (4 KB)
    _Float16* cb16     = (_Float16*)(ws + 4096);           // K*D halfs   (128 KB)
    int*      idx      = (int*)(ws + 4096 + 131072);       // N ints      (512 KB)
    float*    partials = (float*)(ws + 4096 + 131072 + 524288); // ND/256 floats (128 KB)

    float* out      = (float*)d_out;
    float* out_loss = out;                                  // [1]
    float* out_q    = out + 1;                              // [N, D]
    float* out_idx  = out + 1 + (size_t)VQ_ND;              // [N] as float

    vq_prep_kernel<<<(VQ_K + 255) / 256, 256, 0, stream>>>(cb, cb16, cnorm);

    const size_t smem = (size_t)VQ_K * LDS_STRIDE * 2 + (size_t)VQ_K * 4; // 151552 B
    vq_argmin_kernel<<<VQ_N / 128, 256, smem, stream>>>(x, cb16, cnorm, idx, out_idx);

    vq_gather_kernel<<<VQ_ND / 256, 256, 0, stream>>>(x, cb, idx, out_q, partials);

    vq_loss_kernel<<<1, 256, 0, stream>>>(partials, out_loss);
}
